// MGVAE_19310172963482
// MI455X (gfx1250) — matrix operand order repaired and checked by reading, not compiled, not run on hardware
//
#include <hip/hip_runtime.h>

// ---------------- problem constants ----------------
#define BB 8
#define NN 256
#define DN 16
#define DE 8
#define HH 64
#define ZZ 64
#define LL 3
#define FC 128
#define ROWS (BB*NN)          // 2048
#define EDGES (BB*NN*NN)      // 524288

typedef _Float16 v16h __attribute__((ext_vector_type(16)));
typedef float    v8f  __attribute__((ext_vector_type(8)));

// ---------------- device helpers ----------------
__device__ __forceinline__ float fast_tanh(float x) {
  x = fminf(fmaxf(x, -12.0f), 12.0f);
  float e = __expf(2.0f * x);
  return (e - 1.0f) / (e + 1.0f);
}

__device__ __forceinline__ v8f wmma32(v16h a, v16h b, v8f c) {
  // D = A(16x32 f16) * B(32x16 f16) + C(16x16 f32)
  return __builtin_amdgcn_wmma_f32_16x16x32_f16(
      /*neg_a=*/false, a, /*neg_b=*/false, b,
      /*c_mod=*/(short)0, c, /*reuse_a=*/false, /*reuse_b=*/false);
}

// B fragment: weights stored transposed [Nout][Kpad] f16 (global or LDS).
// Lane layout: n = lane&15, lanes 0-15 hold K=k0..k0+15, lanes 16-31 K=k0+16..k0+31.
// 16 contiguous halfs per lane -> 2x b128 loads.
__device__ __forceinline__ v16h load_bfrag(const _Float16* wt, int Kpad, int n0,
                                           int k0, int lane) {
  int n = lane & 15, hf = lane >> 4;
  const _Float16* p = wt + (size_t)(n0 + n) * Kpad + k0 + hf * 16;
  v16h b;
#pragma unroll
  for (int e = 0; e < 16; ++e) b[e] = p[e];
  return b;
}

// A fragment from row-major LDS tile [16][Kpad] f16.
// 16-bit A 16x32, contiguous k (not ISA table order): lane m holds K = k0 + hf*16 + {0..15} (matches load_bfrag).
__device__ __forceinline__ v16h load_afrag(const _Float16* tile, int Kpad, int k0,
                                           int lane) {
  int m = lane & 15, hf = lane >> 4;
  const _Float16* row = tile + (size_t)m * Kpad + k0;
  const _Float16* p0 = row + hf * 16;
  const _Float16* p1 = row + hf * 16 + 8;
  v16h a;
#pragma unroll
  for (int e = 0; e < 8; ++e) { a[e] = p0[e]; a[8 + e] = p1[e]; }
  return a;
}

// ---------------- weight transpose + f16 convert ----------------
// src f32 [K][Nn] row-major  ->  dst f16 [Nn][Kpad] (zero padded)
__global__ void k_transpose_w(const float* __restrict__ src,
                              _Float16* __restrict__ dst, int K, int Nn, int Kpad) {
  int idx = blockIdx.x * 256 + threadIdx.x;
  if (idx >= Nn * Kpad) return;
  int n = idx / Kpad;
  int k = idx - n * Kpad;
  float v = (k < K) ? src[(size_t)k * Nn + n] : 0.0f;
  dst[(size_t)n * Kpad + k] = (_Float16)v;
}

// ---------------- fused edge MLP + channel reduction ----------------
// E[e] = sum_c tanh( tanh(x_e @ W1 + b1) @ W2 + b2 )[c],  x_e in R^8
// One wave handles 16 edges: 8 WMMAs (layer1, K pad 8->32) + 16 WMMAs (layer2 K=128)
__global__ void k_edge_mlp(const float* __restrict__ edge,   // [EDGES][8]
                           const _Float16* __restrict__ w1t, // [128][32]
                           const float* __restrict__ b1,
                           const _Float16* __restrict__ w2t, // [64][128]
                           const float* __restrict__ b2,
                           float* __restrict__ E) {
  __shared__ _Float16 mid[8][16 * 128];
  int wave = threadIdx.x >> 5, lane = threadIdx.x & 31;
  int m = lane & 15, hf = lane >> 4;
  size_t base = ((size_t)blockIdx.x * 8 + wave) * 16;

  // prefetch a future block's edge rows into L2 (global_prefetch_b8)
  __builtin_prefetch(edge + (base + (size_t)8 * 16) * DE, 0, 1);

  // Build A fragment directly: real K=8 lives in half 0, VGPRs 0-3.
  v16h a = {};
  if (hf == 0) {
    const float* row = edge + (base + m) * DE;
#pragma unroll
    for (int e = 0; e < 8; ++e) a[e] = (_Float16)row[e];
  }

  _Float16* md = mid[wave];
  // layer 1: [16x32(pad)] x [32x16] -> 8 N-tiles of FC=128
#pragma unroll
  for (int nt = 0; nt < 8; ++nt) {
    v8f c = {};
    v16h b = load_bfrag(w1t, 32, nt * 16, 0, lane);
    c = wmma32(a, b, c);
    float bv = b1[nt * 16 + m];
#pragma unroll
    for (int r = 0; r < 8; ++r) {
      float v = fast_tanh(c[r] + bv);
      md[(size_t)(r + 8 * hf) * 128 + nt * 16 + m] = (_Float16)v;
    }
  }
  __syncthreads();

  // layer 2: K=128 (4 chunks), OUT=64 (4 N-tiles); per-lane partial column sums
  float sums[8];
#pragma unroll
  for (int r = 0; r < 8; ++r) sums[r] = 0.0f;
#pragma unroll
  for (int nt = 0; nt < 4; ++nt) {
    v8f c = {};
#pragma unroll
    for (int kc = 0; kc < 4; ++kc) {
      v16h aa = load_afrag(md, 128, kc * 32, lane);
      v16h bb = load_bfrag(w2t, 128, nt * 16, kc * 32, lane);
      c = wmma32(aa, bb, c);
    }
    float bv = b2[nt * 16 + m];
#pragma unroll
    for (int r = 0; r < 8; ++r) sums[r] += fast_tanh(c[r] + bv);
  }
  // cross-lane reduce over the 16 columns within each lane half
#pragma unroll
  for (int r = 0; r < 8; ++r) {
    float v = sums[r];
    v += __shfl_xor(v, 1, 32);
    v += __shfl_xor(v, 2, 32);
    v += __shfl_xor(v, 4, 32);
    v += __shfl_xor(v, 8, 32);
    if (m == r) E[base + r + 8 * hf] = v;  // row = r + 8*half
  }
}

// ---------------- generic two-layer MLP: K1 -> 128 -> 64 ----------------
// 128 threads = 4 waves, one 16-row tile per wave. in f32 (row stride istride).
__global__ void k_mlp2(const float* __restrict__ in, int istride, int K1, int K1pad,
                       int kshift, const _Float16* __restrict__ w1t,
                       const float* __restrict__ b1,
                       const _Float16* __restrict__ w2t,
                       const float* __restrict__ b2, int tanh_out,
                       float* __restrict__ out, int ostride) {
  __shared__ _Float16 tin[4][16 * 256];
  __shared__ _Float16 mid[4][16 * 128];
  int wave = threadIdx.x >> 5, lane = threadIdx.x & 31;
  int m = lane & 15, hf = lane >> 4;
  size_t r0 = ((size_t)blockIdx.x * 4 + wave) * 16;
  _Float16* ti = tin[wave];
  _Float16* md = mid[wave];

  for (int idx = lane; idx < 16 * K1pad; idx += 32) {
    int r = idx >> kshift;
    int k = idx & (K1pad - 1);
    float v = (k < K1) ? in[(r0 + r) * (size_t)istride + k] : 0.0f;
    ti[idx] = (_Float16)v;
  }
  __syncthreads();

  int KC1 = K1pad >> 5;
  for (int nt = 0; nt < 8; ++nt) {  // FC = 128
    v8f c = {};
    for (int kc = 0; kc < KC1; ++kc) {
      v16h a = load_afrag(ti, K1pad, kc * 32, lane);
      v16h b = load_bfrag(w1t, K1pad, nt * 16, kc * 32, lane);
      c = wmma32(a, b, c);
    }
    float bv = b1[nt * 16 + m];
#pragma unroll
    for (int r = 0; r < 8; ++r)
      md[(size_t)(r + 8 * hf) * 128 + nt * 16 + m] = (_Float16)fast_tanh(c[r] + bv);
  }
  __syncthreads();

#pragma unroll
  for (int nt = 0; nt < 4; ++nt) {  // OUT = 64
    v8f c = {};
#pragma unroll
    for (int kc = 0; kc < 4; ++kc) {
      v16h a = load_afrag(md, 128, kc * 32, lane);
      v16h b = load_bfrag(w2t, 128, nt * 16, kc * 32, lane);
      c = wmma32(a, b, c);
    }
    float bv = b2[nt * 16 + m];
#pragma unroll
    for (int r = 0; r < 8; ++r) {
      float v = c[r] + bv;
      if (tanh_out) v = fast_tanh(v);
      out[(r0 + r + 8 * hf) * (size_t)ostride + nt * 16 + m] = v;
    }
  }
}

// ---------------- single row-GEMM: [rows x K] @ Wt[64][K] ----------------
__global__ void k_rowgemm(const float* __restrict__ in, int istride, int K,
                          int kshift, const _Float16* __restrict__ wt,
                          const float* __restrict__ bias, int dotanh,
                          float* __restrict__ out, int ostride) {
  __shared__ _Float16 tiles[4][16 * 128];
  int wave = threadIdx.x >> 5, lane = threadIdx.x & 31;
  int m = lane & 15, hf = lane >> 4;
  size_t r0 = ((size_t)blockIdx.x * 4 + wave) * 16;
  _Float16* tl = tiles[wave];

  for (int idx = lane; idx < 16 * K; idx += 32) {
    int r = idx >> kshift;
    int k = idx & (K - 1);
    tl[idx] = (_Float16)in[(r0 + r) * (size_t)istride + k];
  }
  __syncthreads();

  int KC = K >> 5;
#pragma unroll
  for (int nt = 0; nt < 4; ++nt) {
    v8f c = {};
    for (int kc = 0; kc < KC; ++kc) {
      v16h a = load_afrag(tl, K, kc * 32, lane);
      v16h b = load_bfrag(wt, K, nt * 16, kc * 32, lane);
      c = wmma32(a, b, c);
    }
    float bv = bias ? bias[nt * 16 + m] : 0.0f;
#pragma unroll
    for (int r = 0; r < 8; ++r) {
      float v = c[r] + bv;
      if (dotanh) v = fast_tanh(v);
      out[(r0 + r + 8 * hf) * (size_t)ostride + nt * 16 + m] = v;
    }
  }
}

// ---------------- batched OUT = tanh(M[b] @ X[b]) ----------------
// M: [B][256][256] f32, X: [B][256][xstride] (cols 0..63), OUT rows stride ostride.
// grid = B*4; block = 128 threads; each block: 64 rows of one batch.
__global__ void k_bmm_tanh(const float* __restrict__ M, const float* __restrict__ X,
                           int xstride, float* __restrict__ out, int ostride) {
  __shared__ _Float16 XT[64 * 256];  // X^T[n][k] f16
  __shared__ _Float16 Mt[16 * 256];
  int b = blockIdx.x >> 2;
  int quad = blockIdx.x & 3;
  int wave = threadIdx.x >> 5, lane = threadIdx.x & 31;
  int m = lane & 15, hf = lane >> 4;

  const float* Xb = X + (size_t)b * NN * xstride;
  for (int idx = threadIdx.x; idx < 64 * 256; idx += 128) {
    int n = idx >> 8;
    int k = idx & 255;
    XT[idx] = (_Float16)Xb[(size_t)k * xstride + n];
  }
  __syncthreads();

  for (int it = 0; it < 4; ++it) {
    int row0 = quad * 64 + it * 16;
    const float* Mb = M + ((size_t)b * NN + row0) * NN;
    for (int idx = threadIdx.x; idx < 16 * 256; idx += 128)
      Mt[idx] = (_Float16)Mb[idx];
    __syncthreads();

    v8f c = {};
#pragma unroll
    for (int kc = 0; kc < 8; ++kc) {  // K = 256
      v16h a = load_afrag(Mt, 256, kc * 32, lane);
      v16h bb = load_bfrag(XT, 256, wave * 16, kc * 32, lane);
      c = wmma32(a, bb, c);
    }
    float* ob = out + ((size_t)b * NN + row0) * ostride;
#pragma unroll
    for (int r = 0; r < 8; ++r)
      ob[(size_t)(r + 8 * hf) * ostride + wave * 16 + m] = fast_tanh(c[r]);
    __syncthreads();
  }
}

// ---------------- reparameterization + output packing ----------------
__global__ void k_final(const float* __restrict__ noise, const float* __restrict__ mn,
                        const float* __restrict__ lv, float* __restrict__ out) {
  int i = blockIdx.x * 256 + threadIdx.x;  // 131072 total
  float mv = mn[i], l = lv[i];
  out[i] = noise[i] * __expf(0.5f * l) + mv;      // latent
  out[ROWS * ZZ + i] = mv;                        // mean
  out[2 * ROWS * ZZ + i] = l;                     // logvar
}

// ---------------- host launcher ----------------
extern "C" void kernel_launch(void* const* d_in, const int* in_sizes, int n_in,
                              void* d_out, int out_size, void* d_ws, size_t ws_size,
                              hipStream_t stream) {
  (void)in_sizes; (void)n_in; (void)out_size; (void)ws_size;

  const float* adj        = (const float*)d_in[0];
  const float* node_feat  = (const float*)d_in[1];
  const float* edge_feat  = (const float*)d_in[2];
  const float* noise      = (const float*)d_in[3];
  const float* node_fc1_w = (const float*)d_in[4];  const float* node_fc1_b = (const float*)d_in[5];
  const float* node_fc2_w = (const float*)d_in[6];  const float* node_fc2_b = (const float*)d_in[7];
  const float* edge_fc1_w = (const float*)d_in[8];  const float* edge_fc1_b = (const float*)d_in[9];
  const float* edge_fc2_w = (const float*)d_in[10]; const float* edge_fc2_b = (const float*)d_in[11];
  const float* gcn_w      = (const float*)d_in[12];
  const float* comb_w     = (const float*)d_in[13];
  const float* comb_b     = (const float*)d_in[14];
  const float* mean_fc1_w = (const float*)d_in[15]; const float* mean_fc1_b = (const float*)d_in[16];
  const float* mean_fc2_w = (const float*)d_in[17]; const float* mean_fc2_b = (const float*)d_in[18];
  const float* lv_fc1_w   = (const float*)d_in[19]; const float* lv_fc1_b   = (const float*)d_in[20];
  const float* lv_fc2_w   = (const float*)d_in[21]; const float* lv_fc2_b   = (const float*)d_in[22];

  // ---- workspace carving ----
  char* ws = (char*)d_ws;
  size_t off = 0;
  auto carve = [&](size_t bytes) -> char* {
    char* p = ws + off;
    off = (off + bytes + 255) & ~(size_t)255;
    return p;
  };
  _Float16* wt_node1 = (_Float16*)carve(128 * 32 * 2);
  _Float16* wt_node2 = (_Float16*)carve(64 * 128 * 2);
  _Float16* wt_edge1 = (_Float16*)carve(128 * 32 * 2);
  _Float16* wt_edge2 = (_Float16*)carve(64 * 128 * 2);
  _Float16* wt_gcn   = (_Float16*)carve((size_t)LL * 64 * 64 * 2);
  _Float16* wt_comb  = (_Float16*)carve((size_t)LL * 64 * 128 * 2);
  _Float16* wt_mean1 = (_Float16*)carve(128 * 256 * 2);
  _Float16* wt_mean2 = (_Float16*)carve(64 * 128 * 2);
  _Float16* wt_lv1   = (_Float16*)carve(128 * 256 * 2);
  _Float16* wt_lv2   = (_Float16*)carve(64 * 128 * 2);
  float* E     = (float*)carve((size_t)EDGES * 4);        // [B][N][N]
  float* hcat  = (float*)carve((size_t)ROWS * 256 * 4);   // [2048][256]
  float* hc    = (float*)carve((size_t)ROWS * 128 * 4);   // [2048][128] = [h | comb]
  float* tbuf  = (float*)carve((size_t)ROWS * 64 * 4);
  float* meanb = (float*)carve((size_t)ROWS * 64 * 4);
  float* lvb   = (float*)carve((size_t)ROWS * 64 * 4);

  // ---- weight pre-transpose/convert (f32 [K][N] -> f16 [N][Kpad]) ----
  auto T = [&](const float* s, _Float16* d, int K, int Nn, int Kpad) {
    int total = Nn * Kpad;
    k_transpose_w<<<(total + 255) / 256, 256, 0, stream>>>(s, d, K, Nn, Kpad);
  };
  T(node_fc1_w, wt_node1, 16, 128, 32);
  T(node_fc2_w, wt_node2, 128, 64, 128);
  T(edge_fc1_w, wt_edge1, 8, 128, 32);
  T(edge_fc2_w, wt_edge2, 128, 64, 128);
  for (int l = 0; l < LL; ++l) T(gcn_w + (size_t)l * 64 * 64, wt_gcn + (size_t)l * 64 * 64, 64, 64, 64);
  for (int l = 0; l < LL; ++l) T(comb_w + (size_t)l * 128 * 64, wt_comb + (size_t)l * 64 * 128, 128, 64, 128);
  T(mean_fc1_w, wt_mean1, 256, 128, 256);
  T(mean_fc2_w, wt_mean2, 128, 64, 128);
  T(lv_fc1_w, wt_lv1, 256, 128, 256);
  T(lv_fc2_w, wt_lv2, 128, 64, 128);

  // ---- node embedding -> hcat[:, 0:64] ----
  k_mlp2<<<ROWS / 64, 128, 0, stream>>>(node_feat, DN, 16, 32, 5, wt_node1,
                                        node_fc1_b, wt_node2, node_fc2_b,
                                        /*tanh_out=*/1, hcat, 256);

  // ---- fused edge MLP + channel sum -> E ----
  k_edge_mlp<<<EDGES / 128, 256, 0, stream>>>(edge_feat, wt_edge1, edge_fc1_b,
                                              wt_edge2, edge_fc2_b, E);

  // ---- GCN layers ----
  for (int l = 0; l < LL; ++l) {
    // t = hidden_l @ gcn_w[l]            (no bias, no tanh)
    k_rowgemm<<<ROWS / 64, 128, 0, stream>>>(hcat + l * 64, 256, 64, 6,
                                             wt_gcn + (size_t)l * 64 * 64, nullptr,
                                             0, tbuf, 64);
    // h = tanh(adj @ t)      -> hc[:, 0:64]
    k_bmm_tanh<<<BB * 4, 128, 0, stream>>>(adj, tbuf, 64, hc, 128);
    // comb = tanh(E @ h)     -> hc[:, 64:128]
    k_bmm_tanh<<<BB * 4, 128, 0, stream>>>(E, hc, 128, hc + 64, 128);
    // hidden_{l+1} = tanh(hc @ comb_w[l] + comb_b[l]) -> hcat[:, (l+1)*64:...]
    k_rowgemm<<<ROWS / 64, 128, 0, stream>>>(hc, 128, 128, 7,
                                             wt_comb + (size_t)l * 64 * 128,
                                             comb_b + l * 64, 1,
                                             hcat + (l + 1) * 64, 256);
  }

  // ---- heads ----
  k_mlp2<<<ROWS / 64, 128, 0, stream>>>(hcat, 256, 256, 256, 8, wt_mean1,
                                        mean_fc1_b, wt_mean2, mean_fc2_b,
                                        /*tanh_out=*/0, meanb, 64);
  k_mlp2<<<ROWS / 64, 128, 0, stream>>>(hcat, 256, 256, 256, 8, wt_lv1, lv_fc1_b,
                                        wt_lv2, lv_fc2_b, /*tanh_out=*/0, lvb, 64);

  // ---- latent + pack outputs ----
  k_final<<<(ROWS * ZZ) / 256, 256, 0, stream>>>(noise, meanb, lvb, (float*)d_out);
}
